// TransformerBlock_24361054502930
// MI455X (gfx1250) — matrix operand order repaired and checked by reading, not compiled, not run on hardware
//
#include <hip/hip_runtime.h>
#include <hip/hip_bf16.h>

// ---------------------------------------------------------------------------
// Transformer block forward for gfx1250 (MI455X), bf16 WMMA + fp32 accumulate.
// ---------------------------------------------------------------------------

#define B_  2
#define S_  2048
#define D_  1024
#define H_  16
#define DK_ 64
#define DFF_ 4096
#define MROWS (B_ * S_)   // 4096

typedef __attribute__((ext_vector_type(16))) __bf16 bf16x16;
typedef __attribute__((ext_vector_type(8)))  __bf16 bf16x8;
typedef __attribute__((ext_vector_type(8)))  float  f32x8;

__device__ __forceinline__ f32x8 zero_f32x8() {
    f32x8 z = {0.f, 0.f, 0.f, 0.f, 0.f, 0.f, 0.f, 0.f};
    return z;
}

__device__ __forceinline__ bf16x16 join8(bf16x8 lo, bf16x8 hi) {
    return __builtin_shufflevector(lo, hi, 0,1,2,3,4,5,6,7,8,9,10,11,12,13,14,15);
}

__device__ __forceinline__ f32x8 wmma_bf16(bf16x16 a, bf16x16 b, f32x8 c) {
    // D = A(16x32 bf16) * B(32x16 bf16) + C(16x16 f32)
    return __builtin_amdgcn_wmma_f32_16x16x32_bf16(false, a, false, b, (short)0, c, false, false);
}

// A fragment (16x32, row-major source [.,ld]): lane r=row, K segs [half*8,+8) and [16+half*8,+8)
__device__ __forceinline__ bf16x16 load_a_frag(const __bf16* __restrict__ base, int row0,
                                               int k0, int ld, int r, int half) {
    const __bf16* p = base + (size_t)(row0 + r) * ld + k0 + half * 8;
    return join8(*(const bf16x8*)p, *(const bf16x8*)(p + 16));
}

// B fragment (32x16) from a row-major [N,K] "transposed weight": lane r=col, K segs [half*8,+8) and [16+half*8,+8)
__device__ __forceinline__ bf16x16 load_b_frag(const __bf16* __restrict__ baseT, int col0,
                                               int k0, int ld, int r, int half) {
    const __bf16* p = baseT + (size_t)(col0 + r) * ld + k0 + half * 8;
    return join8(*(const bf16x8*)p, *(const bf16x8*)(p + 16));
}

// ---------------------------------------------------------------------------
// fp32 -> bf16 elementwise convert
// ---------------------------------------------------------------------------
__global__ __launch_bounds__(256) void cvt_bf16_kernel(const float* __restrict__ in,
                                                       __bf16* __restrict__ out, int n) {
    for (int i = blockIdx.x * blockDim.x + threadIdx.x; i < n; i += gridDim.x * blockDim.x)
        out[i] = (__bf16)in[i];
}

// ---------------------------------------------------------------------------
// fp32 [R][C] -> bf16 [C][R] transpose+convert (LDS tiled, 32x32)
// ---------------------------------------------------------------------------
__global__ __launch_bounds__(256) void transpose_cvt_kernel(const float* __restrict__ in,
                                                            __bf16* __restrict__ out,
                                                            int R, int C) {
    __shared__ float tile[32][33];
    int c0 = blockIdx.x * 32, r0 = blockIdx.y * 32;
    int tx = threadIdx.x & 31, ty = threadIdx.x >> 5;   // 32 x 8
#pragma unroll
    for (int j = 0; j < 32; j += 8)
        tile[ty + j][tx] = in[(size_t)(r0 + ty + j) * C + c0 + tx];
    __syncthreads();
#pragma unroll
    for (int j = 0; j < 32; j += 8)
        out[(size_t)(c0 + ty + j) * R + r0 + tx] = (__bf16)tile[tx][ty + j];
}

// ---------------------------------------------------------------------------
// Generic GEMM: out[M,N] = A[M,K](bf16) @ Bt[N,K]^T(bf16) + bias
// 8 waves/block, wave tile 32x64, block tile 128x128, K step 32.
// Fragments come straight from global (L2-resident on 192MB L2).
// ---------------------------------------------------------------------------
__global__ __launch_bounds__(256) void gemm_wmma_kernel(
    const __bf16* __restrict__ A, const __bf16* __restrict__ Bt,
    const float* __restrict__ bias, float* __restrict__ outF, __bf16* __restrict__ outB,
    int M, int N, int K, int relu, int vtrans, int Svt) {
    int tid = threadIdx.x, lane = tid & 31, wave = tid >> 5;
    int r = lane & 15, half = lane >> 4;
    int m0 = blockIdx.y * 128 + (wave >> 1) * 32;
    int n0 = blockIdx.x * 128 + (wave & 1) * 64;

    f32x8 acc[2][4];
#pragma unroll
    for (int mc = 0; mc < 2; ++mc)
#pragma unroll
        for (int nc = 0; nc < 4; ++nc) acc[mc][nc] = zero_f32x8();

    for (int k0 = 0; k0 < K; k0 += 32) {
        bf16x16 a0 = load_a_frag(A, m0,      k0, K, r, half);
        bf16x16 a1 = load_a_frag(A, m0 + 16, k0, K, r, half);
#pragma unroll
        for (int nc = 0; nc < 4; ++nc) {
            bf16x16 bfr = load_b_frag(Bt, n0 + nc * 16, k0, K, r, half);
            acc[0][nc] = wmma_bf16(a0, bfr, acc[0][nc]);
            acc[1][nc] = wmma_bf16(a1, bfr, acc[1][nc]);
        }
    }

#pragma unroll
    for (int mc = 0; mc < 2; ++mc)
#pragma unroll
        for (int nc = 0; nc < 4; ++nc)
#pragma unroll
            for (int j = 0; j < 8; ++j) {
                int m = m0 + mc * 16 + j + (half ? 8 : 0);
                int n = n0 + nc * 16 + r;
                float v = acc[mc][nc][j];
                if (bias) v += bias[n];
                if (relu) v = fmaxf(v, 0.0f);
                if (outF) outF[(size_t)m * N + n] = v;
                if (outB) {
                    if (vtrans) {  // store as [b][N][Svt] (V transposed for attention B-frags)
                        int bb = m / Svt, ss = m - bb * Svt;
                        outB[((size_t)bb * N + n) * Svt + ss] = (__bf16)v;
                    } else {
                        outB[(size_t)m * N + n] = (__bf16)v;
                    }
                }
            }
}

// ---------------------------------------------------------------------------
// Fused attention with softmax over the HEADS axis.
// One block per (batch, 16-row q-tile); 16 waves, wave h == head h.
// Per 32-key tile: scores via WMMA -> LDS [q][k][h] -> cooperative softmax
// across the 16 contiguous head values -> bf16 attn in A-frag layout ->
// ctx accumulation via WMMA against V^T.
// ---------------------------------------------------------------------------
__global__ __launch_bounds__(512) void attn_heads_softmax_kernel(
    const __bf16* __restrict__ qb, const __bf16* __restrict__ kb,
    const __bf16* __restrict__ vT, __bf16* __restrict__ ctxb) {
    __shared__ __align__(16) float  s_sc[16 * 32 * 16];   // [q][k][h] fp32   (32 KB)
    __shared__ __align__(16) __bf16 s_at[16 * 16 * 32];   // [h][q][k] bf16   (16 KB)

    int tid = threadIdx.x, lane = tid & 31, h = tid >> 5;
    int r = lane & 15, half = lane >> 4;
    int b  = blockIdx.x / (S_ / 16);
    int q0 = (blockIdx.x % (S_ / 16)) * 16;

    // Q fragments for this head / q-tile (dk=64 -> two 16x32 A-frags), loaded once.
    bf16x16 qf[2];
#pragma unroll
    for (int c = 0; c < 2; ++c) {
        const __bf16* p = qb + (size_t)(b * S_ + q0 + r) * D_ + h * DK_ + c * 32 + half * 8;
        qf[c] = join8(*(const bf16x8*)p, *(const bf16x8*)(p + 16));
    }

    f32x8 ctx[4];
#pragma unroll
    for (int dc = 0; dc < 4; ++dc) ctx[dc] = zero_f32x8();

    for (int kt = 0; kt < S_ / 32; ++kt) {
        int k0 = kt * 32;
        // --- scores for this head over a 16q x 32k tile ---
#pragma unroll
        for (int kk = 0; kk < 2; ++kk) {
            f32x8 sc = zero_f32x8();
#pragma unroll
            for (int c = 0; c < 2; ++c) {
                // B-frag = K^T chunk: lane = key column, dk segs [half*8,+8) and [16+half*8,+8)
                const __bf16* p = kb + (size_t)(b * S_ + k0 + kk * 16 + r) * D_
                                     + h * DK_ + c * 32 + half * 8;
                bf16x16 bfr = join8(*(const bf16x8*)p, *(const bf16x8*)(p + 16));
                sc = wmma_bf16(qf[c], bfr, sc);
            }
#pragma unroll
            for (int j = 0; j < 8; ++j) {
                int ql = j + (half ? 8 : 0);
                int kc = kk * 16 + r;
                s_sc[(ql * 32 + kc) * 16 + h] = sc[j] * 0.125f;  // 1/sqrt(64)
            }
        }
        __syncthreads();

        // --- cooperative softmax across heads: 512 threads x 512 (q,k) cells ---
        {
            int q = tid >> 5, kc = tid & 31;
            const float* p = &s_sc[(q * 32 + kc) * 16];
            float mx = p[0];
#pragma unroll
            for (int i = 1; i < 16; ++i) mx = fmaxf(mx, p[i]);
            float e[16], sum = 0.f;
#pragma unroll
            for (int i = 0; i < 16; ++i) { e[i] = __expf(p[i] - mx); sum += e[i]; }
            float rs = 1.0f / sum;
#pragma unroll
            for (int i = 0; i < 16; ++i)
                s_at[i * 512 + q * 32 + kc] = (__bf16)(e[i] * rs);
        }
        __syncthreads();

        // --- ctx += attn(16x32) @ V(32x64) for this head ---
        const __bf16* ap = &s_at[h * 512 + r * 32];
        bf16x16 af = join8(*(const bf16x8*)(ap + half * 8),
                           *(const bf16x8*)(ap + 16 + half * 8));
#pragma unroll
        for (int dc = 0; dc < 4; ++dc) {
            const __bf16* p = vT + (size_t)b * D_ * S_
                                 + (size_t)(h * DK_ + dc * 16 + r) * S_ + k0 + half * 8;
            bf16x16 vf = join8(*(const bf16x8*)p, *(const bf16x8*)(p + 16));
            ctx[dc] = wmma_bf16(af, vf, ctx[dc]);
        }
    }

    // store ctx (bf16) back in [B*S, D] layout for the O-projection GEMM
#pragma unroll
    for (int dc = 0; dc < 4; ++dc)
#pragma unroll
        for (int j = 0; j < 8; ++j) {
            int m = b * S_ + q0 + j + (half ? 8 : 0);
            ctxb[(size_t)m * D_ + h * DK_ + dc * 16 + r] = (__bf16)ctx[dc][j];
        }
}

// ---------------------------------------------------------------------------
// Fused residual-add + LayerNorm. One block per row (D=1024, 256 threads x4).
// ---------------------------------------------------------------------------
__global__ __launch_bounds__(256) void ln_kernel(const float* __restrict__ a,
                                                 const float* __restrict__ bres,
                                                 const float* __restrict__ g,
                                                 const float* __restrict__ be,
                                                 float* __restrict__ outF,
                                                 __bf16* __restrict__ outB) {
    int row = blockIdx.x, tid = threadIdx.x;
    const float* pa = a + (size_t)row * D_;
    const float* pb = bres + (size_t)row * D_;
    float v[4], s = 0.f;
#pragma unroll
    for (int i = 0; i < 4; ++i) { int n = tid + i * 256; v[i] = pa[n] + pb[n]; s += v[i]; }

    __shared__ float red[8];
#pragma unroll
    for (int o = 16; o > 0; o >>= 1) s += __shfl_xor(s, o, 32);
    if ((tid & 31) == 0) red[tid >> 5] = s;
    __syncthreads();
    float tot = 0.f;
#pragma unroll
    for (int w = 0; w < 8; ++w) tot += red[w];
    float mean = tot * (1.0f / D_);
    __syncthreads();

    float vs = 0.f;
#pragma unroll
    for (int i = 0; i < 4; ++i) { float d = v[i] - mean; vs += d * d; }
#pragma unroll
    for (int o = 16; o > 0; o >>= 1) vs += __shfl_xor(vs, o, 32);
    if ((tid & 31) == 0) red[tid >> 5] = vs;
    __syncthreads();
    float vt = 0.f;
#pragma unroll
    for (int w = 0; w < 8; ++w) vt += red[w];
    float rstd = rsqrtf(vt * (1.0f / D_) + 1e-5f);

#pragma unroll
    for (int i = 0; i < 4; ++i) {
        int n = tid + i * 256;
        float y = (v[i] - mean) * rstd * g[n] + be[n];
        if (outF) outF[(size_t)row * D_ + n] = y;
        if (outB) outB[(size_t)row * D_ + n] = (__bf16)y;
    }
}

// ---------------------------------------------------------------------------
// Host orchestration
// ---------------------------------------------------------------------------
extern "C" void kernel_launch(void* const* d_in, const int* in_sizes, int n_in,
                              void* d_out, int out_size, void* d_ws, size_t ws_size,
                              hipStream_t stream) {
    (void)in_sizes; (void)n_in; (void)out_size; (void)ws_size;
    const float* x     = (const float*)d_in[0];
    const float* wq    = (const float*)d_in[1];
    const float* bq    = (const float*)d_in[2];
    const float* wk    = (const float*)d_in[3];
    const float* bk    = (const float*)d_in[4];
    const float* wv    = (const float*)d_in[5];
    const float* bv    = (const float*)d_in[6];
    const float* wo    = (const float*)d_in[7];
    const float* bo    = (const float*)d_in[8];
    const float* g1    = (const float*)d_in[9];
    const float* beta1 = (const float*)d_in[10];
    const float* w1    = (const float*)d_in[11];
    const float* b1    = (const float*)d_in[12];
    const float* w2    = (const float*)d_in[13];
    const float* b2    = (const float*)d_in[14];
    const float* g2    = (const float*)d_in[15];
    const float* beta2 = (const float*)d_in[16];

    char* ws = (char*)d_ws;
    size_t off = 0;
    auto alloc = [&](size_t bytes) -> void* {
        off = (off + 255) & ~(size_t)255;
        void* p = ws + off;
        off += bytes;
        return p;
    };

    const size_t nXD  = (size_t)MROWS * D_;
    const size_t nXF  = (size_t)MROWS * DFF_;

    __bf16* xb   = (__bf16*)alloc(nXD * 2);
    __bf16* wqT  = (__bf16*)alloc((size_t)D_ * D_ * 2);
    __bf16* wkT  = (__bf16*)alloc((size_t)D_ * D_ * 2);
    __bf16* wvT  = (__bf16*)alloc((size_t)D_ * D_ * 2);
    __bf16* woT  = (__bf16*)alloc((size_t)D_ * D_ * 2);
    __bf16* w1T  = (__bf16*)alloc((size_t)DFF_ * D_ * 2);
    __bf16* w2T  = (__bf16*)alloc((size_t)D_ * DFF_ * 2);
    __bf16* qbuf = (__bf16*)alloc(nXD * 2);
    __bf16* kbuf = (__bf16*)alloc(nXD * 2);
    __bf16* vT   = (__bf16*)alloc(nXD * 2);
    __bf16* ctxb = (__bf16*)alloc(nXD * 2);
    float*  aout = (float*)alloc(nXD * 4);
    float*  x1f  = (float*)alloc(nXD * 4);
    __bf16* x1b  = (__bf16*)alloc(nXD * 2);
    __bf16* h1b  = (__bf16*)alloc(nXF * 2);
    float*  fff  = (float*)alloc(nXD * 4);

    // 1) convert x, transpose+convert all weights to bf16 [N,K]
    cvt_bf16_kernel<<<4096, 256, 0, stream>>>(x, xb, (int)nXD);
    transpose_cvt_kernel<<<dim3(D_ / 32, D_ / 32), 256, 0, stream>>>(wq, wqT, D_, D_);
    transpose_cvt_kernel<<<dim3(D_ / 32, D_ / 32), 256, 0, stream>>>(wk, wkT, D_, D_);
    transpose_cvt_kernel<<<dim3(D_ / 32, D_ / 32), 256, 0, stream>>>(wv, wvT, D_, D_);
    transpose_cvt_kernel<<<dim3(D_ / 32, D_ / 32), 256, 0, stream>>>(wo, woT, D_, D_);
    transpose_cvt_kernel<<<dim3(DFF_ / 32, D_ / 32), 256, 0, stream>>>(w1, w1T, D_, DFF_);
    transpose_cvt_kernel<<<dim3(D_ / 32, DFF_ / 32), 256, 0, stream>>>(w2, w2T, DFF_, D_);

    // 2) QKV projections (V stored transposed per batch for attention B-frags)
    dim3 gD(D_ / 128, MROWS / 128);
    gemm_wmma_kernel<<<gD, 256, 0, stream>>>(xb, wqT, bq, nullptr, qbuf, MROWS, D_, D_, 0, 0, 0);
    gemm_wmma_kernel<<<gD, 256, 0, stream>>>(xb, wkT, bk, nullptr, kbuf, MROWS, D_, D_, 0, 0, 0);
    gemm_wmma_kernel<<<gD, 256, 0, stream>>>(xb, wvT, bv, nullptr, vT,   MROWS, D_, D_, 0, 1, S_);

    // 3) attention with softmax over heads
    attn_heads_softmax_kernel<<<B_ * (S_ / 16), 512, 0, stream>>>(qbuf, kbuf, vT, ctxb);

    // 4) O projection -> fp32 (residual consumed by LN1)
    gemm_wmma_kernel<<<gD, 256, 0, stream>>>(ctxb, woT, bo, aout, nullptr, MROWS, D_, D_, 0, 0, 0);

    // 5) LN1(x + attn_out) -> x1 (fp32 for residual, bf16 for FFN GEMM)
    ln_kernel<<<MROWS, 256, 0, stream>>>(x, aout, g1, beta1, x1f, x1b);

    // 6) FFN: relu(x1@w1+b1) @ w2 + b2
    gemm_wmma_kernel<<<dim3(DFF_ / 128, MROWS / 128), 256, 0, stream>>>(
        x1b, w1T, b1, nullptr, h1b, MROWS, DFF_, D_, 1, 0, 0);
    gemm_wmma_kernel<<<gD, 256, 0, stream>>>(h1b, w2T, b2, fff, nullptr, MROWS, D_, DFF_, 0, 0, 0);

    // 7) LN2(x1 + ff) -> output (fp32)
    ln_kernel<<<MROWS, 256, 0, stream>>>(x1f, fff, g2, beta2, (float*)d_out, nullptr);
}
